// SingleHeadAttention_4346506904189
// MI455X (gfx1250) — matrix operand order repaired and checked by reading, not compiled, not run on hardware
//
#include <hip/hip_runtime.h>
#include <hip/hip_bf16.h>

#define BB 4
#define SS 4096
#define EE 1024
#define HH 64

typedef __attribute__((ext_vector_type(16))) __bf16 bf16x16;
typedef __attribute__((ext_vector_type(8)))  __bf16 bf16x8;
typedef __attribute__((ext_vector_type(8)))  float  f32x8;
typedef __attribute__((ext_vector_type(4)))  unsigned int u32x4;

union ABf {
    bf16x16 v;
    u32x4   u[2];
};

static __device__ inline f32x8 wmma_bf16(bf16x16 a, bf16x16 b, f32x8 c) {
    // emits v_wmma_f32_16x16x32_bf16
    return __builtin_amdgcn_wmma_f32_16x16x32_bf16(false, a, false, b,
                                                   (short)0, c, false, false);
}

// CDNA5 async global->LDS copy (ASYNCcnt-tracked); builtins don't cover these,
// so use inline asm per CDNA5 ISA. The VDST VGPR holds the
// wave-relative LDS byte address == low 32 bits of a generic pointer to LDS.
static __device__ inline void async_copy_b128(unsigned lds_addr, const void* gaddr) {
    asm volatile("global_load_async_to_lds_b128 %0, %1, off"
                 :: "v"(lds_addr), "v"(gaddr) : "memory");
}
static __device__ inline void wait_async0() {
    asm volatile("s_wait_asynccnt 0x0" ::: "memory");
}

// load 8+8 f32 and convert into the WMMA 16-bit A-matrix lane layout
static __device__ inline void load_a_bf16(ABf& a, const float* xp) {
    const float4 f0 = *(const float4*)(xp +  0);
    const float4 f1 = *(const float4*)(xp +  4);
    const float4 f2 = *(const float4*)(xp + 16);
    const float4 f3 = *(const float4*)(xp + 20);
    a.v[0]  = (__bf16)f0.x; a.v[1]  = (__bf16)f0.y;
    a.v[2]  = (__bf16)f0.z; a.v[3]  = (__bf16)f0.w;
    a.v[4]  = (__bf16)f1.x; a.v[5]  = (__bf16)f1.y;
    a.v[6]  = (__bf16)f1.z; a.v[7]  = (__bf16)f1.w;
    a.v[8]  = (__bf16)f2.x; a.v[9]  = (__bf16)f2.y;
    a.v[10] = (__bf16)f2.z; a.v[11] = (__bf16)f2.w;
    a.v[12] = (__bf16)f3.x; a.v[13] = (__bf16)f3.y;
    a.v[14] = (__bf16)f3.z; a.v[15] = (__bf16)f3.w;
}

// ---------------------------------------------------------------------------
// Kernel 0: convert W [1024,64] f32 -> Wt [64,1024] bf16 (transposed), so the
// GEMM B-operand (a column of W) is two contiguous 16B reads in WMMA B layout.
// ---------------------------------------------------------------------------
__global__ __launch_bounds__(256) void wconv_kernel(
    const float* __restrict__ Wq, const float* __restrict__ Wk,
    const float* __restrict__ Wv,
    __bf16* __restrict__ wtq, __bf16* __restrict__ wtk, __bf16* __restrict__ wtv)
{
    const int idx = blockIdx.x * 256 + threadIdx.x;   // 3 * 65536 elements
    const int sel = idx >> 16;
    const int f   = idx & 0xFFFF;                     // flat (k*64 + n)
    const int k   = f >> 6;
    const int n   = f & 63;
    const float*  W  = (sel == 0) ? Wq  : (sel == 1) ? Wk  : Wv;
    __bf16*       Wt = (sel == 0) ? wtq : (sel == 1) ? wtk : wtv;
    Wt[(size_t)n * EE + k] = (__bf16)W[f];
}

// ---------------------------------------------------------------------------
// Kernel 1: fused QKV projection, no LDS, no in-loop barriers.
//   x [B*S,1024] f32 (converted to bf16 A-layout in registers) times
//   Wt [64,1024] bf16 (contiguous B-operand reads, L2-resident).
//   Q,K stored [B*S][64] bf16; V stored transposed [B][64][S] bf16.
//   launch_bounds(.,1) frees VGPRs so 4 B tiles + pipelined A stay live ->
//   4 WMMAs issue back-to-back behind one staged wait.
// ---------------------------------------------------------------------------
__global__ __launch_bounds__(128, 1) void qkv_proj_kernel(
    const float* __restrict__ x,
    const __bf16* __restrict__ wtq, const __bf16* __restrict__ wtk,
    const __bf16* __restrict__ wtv,
    const float* __restrict__ bq, const float* __restrict__ bk,
    const float* __restrict__ bv,
    __bf16* __restrict__ qws, __bf16* __restrict__ kws,
    __bf16* __restrict__ vtws)
{
    const int tid  = threadIdx.x;
    const int lane = tid & 31;
    const int wv   = tid >> 5;
    const int l15  = lane & 15;
    const int hi   = lane >> 4;
    const int sel  = blockIdx.y;

    const __bf16* Wt   = (sel == 0) ? wtq : (sel == 1) ? wtk : wtv;
    const float*  bias = (sel == 0) ? bq  : (sel == 1) ? bk  : bv;

    const int rowbase = blockIdx.x * 64 + wv * 16;
    const int arow    = rowbase + l15;        // this lane's A-matrix row
    const int off     = hi ? 8 : 0;           // A-layout K-offset per lane half

    f32x8 acc[4];
#pragma unroll
    for (int t = 0; t < 4; ++t)
#pragma unroll
        for (int j = 0; j < 8; ++j) acc[t][j] = 0.0f;

    const float*  xrow = x + (size_t)arow * EE + off;
    const __bf16* wrow = Wt + (size_t)l15 * EE + hi * 8;

    // software pipeline: A chunk for k0 is converted one iteration ahead so
    // the cvt of the next chunk never WAR-overlaps the in-flight WMMA's A regs
    ABf a_cur;
    load_a_bf16(a_cur, xrow);

#pragma unroll 4
    for (int k0 = 0; k0 < EE; k0 += 32) {
        // B operands: 4 distinct register sets (independent loads)
        ABf bm[4];
#pragma unroll
        for (int t = 0; t < 4; ++t) {
            const __bf16* wp = wrow + (size_t)t * 16 * EE + k0;
            bm[t].u[0] = *(const u32x4*)(wp);
            bm[t].u[1] = *(const u32x4*)(wp + 16);
        }
        ABf a_next;
        if (k0 + 32 < EE) load_a_bf16(a_next, xrow + k0 + 32);
        // 4 independent WMMAs, back-to-back
#pragma unroll
        for (int t = 0; t < 4; ++t)
            acc[t] = wmma_bf16(a_cur.v, bm[t].v, acc[t]);
        a_cur = a_next;
    }

    // ---- bias + bf16 store (sel branch hoisted: block-uniform) -------------
    if (sel == 2) {
        // Transposed V: a lane's 8 rows are CONSECUTIVE s positions -> pack
        // into a single 16B store per tile.
        const int r0   = rowbase + 8 * hi;
        const int bidx = r0 >> 12;
        const int s0   = r0 & (SS - 1);
#pragma unroll
        for (int t = 0; t < 4; ++t) {
            const int   col = t * 16 + l15;
            const float bs  = bias[col];
            bf16x8 pk;
#pragma unroll
            for (int j = 0; j < 8; ++j) pk[j] = (__bf16)(acc[t][j] + bs);
            *(bf16x8*)(vtws + ((size_t)bidx * HH + col) * SS + s0) = pk;
        }
    } else {
        __bf16* dst = (sel == 0) ? qws : kws;
#pragma unroll
        for (int t = 0; t < 4; ++t) {
            const int   col = t * 16 + l15;
            const float bs  = bias[col];
#pragma unroll
            for (int j = 0; j < 8; ++j) {
                const int r = rowbase + j + 8 * hi;   // C layout: M=j+8*hi
                dst[(size_t)r * HH + col] = (__bf16)(acc[t][j] + bs);
            }
        }
    }
}

// ---------------------------------------------------------------------------
// Kernel 2: flash attention (causal, online softmax, all matmuls via WMMA).
//   Block = 4 waves = 64 query rows. K/V tiles staged into LDS once per block
//   with async global->LDS copies, double-buffered so the copy of tile kt+1
//   overlaps the 16 WMMAs + softmax of tile kt.
// ---------------------------------------------------------------------------
__global__ __launch_bounds__(128, 1) void flash_attn_kernel(
    const __bf16* __restrict__ qws, const __bf16* __restrict__ kws,
    const __bf16* __restrict__ vtws, float* __restrict__ out)
{
    __shared__ __bf16 ktile[2][64 * 64];   // [key][h]  8KB x2
    __shared__ __bf16 vtile[2][64 * 64];   // [h][key]  8KB x2
    __shared__ __bf16 plds[4][16][64];     // per-wave P relayout, 8KB

    const int tid  = threadIdx.x;
    const int lane = tid & 31;
    const int wv   = tid >> 5;
    const int l15  = lane & 15;
    const int hi   = lane >> 4;
    const int b    = blockIdx.y;
    const int qt   = blockIdx.x;
    const int qrowbase = qt * 64 + wv * 16;
    const int off  = hi ? 8 : 0;

    // ---- stage one 64-key K tile + V tile into LDS (async, B128) ----------
    auto stage_tiles = [&](int kb, int buf) {
        // K tile: keys kb..kb+63 are one contiguous 8KB block of kws
        const char* kg   = (const char*)(kws + ((size_t)b * SS + kb) * HH);
        unsigned    klds = (unsigned)(size_t)&ktile[buf][0];
        unsigned    vlds = (unsigned)(size_t)&vtile[buf][0];
#pragma unroll
        for (int i = 0; i < 4; ++i) {
            const int c = tid + i * 128;            // 16B chunk id, 0..511
            async_copy_b128(klds + c * 16, kg + c * 16);
        }
        // V tile: 64 rows (h) of 128B each from transposed V
#pragma unroll
        for (int i = 0; i < 4; ++i) {
            const int c  = tid + i * 128;
            const int h  = c >> 3;
            const int kp = (c & 7) * 8;
            const __bf16* vg = vtws + ((size_t)b * HH + h) * SS + kb + kp;
            async_copy_b128(vlds + c * 16, vg);
        }
    };

    // ---- Q resident in A-layout registers (2 K-steps over H=64) ------------
    ABf qa[2];
    {
        const __bf16* qp = qws + ((size_t)(b * SS) + qrowbase + l15) * HH;
#pragma unroll
        for (int ks = 0; ks < 2; ++ks) {
            qa[ks].u[0] = *(const u32x4*)(qp + ks * 32 + off);
            qa[ks].u[1] = *(const u32x4*)(qp + ks * 32 + off + 16);
        }
    }

    f32x8 acc[4];
    float m[8], lsum[8];
#pragma unroll
    for (int t = 0; t < 4; ++t)
#pragma unroll
        for (int j = 0; j < 8; ++j) acc[t][j] = 0.0f;
#pragma unroll
    for (int j = 0; j < 8; ++j) { m[j] = -1e30f; lsum[j] = 0.0f; }

    // prologue: stage tile 0, wait, sync
    stage_tiles(0, 0);
    wait_async0();
    __syncthreads();

    for (int kt = 0; kt <= qt; ++kt) {
        const int cur = kt & 1;
        if (kt < qt) stage_tiles((kt + 1) * 64, cur ^ 1);   // overlap next tile

        const __bf16* kcur = &ktile[cur][0];
        const __bf16* vcur = &vtile[cur][0];

        // ---- scores: S = Q * K^T from LDS ----------------------------------
        f32x8 c[4];
#pragma unroll
        for (int t = 0; t < 4; ++t)
#pragma unroll
            for (int j = 0; j < 8; ++j) c[t][j] = 0.0f;

#pragma unroll
        for (int ks = 0; ks < 2; ++ks) {
            ABf bm[4];                         // 4 distinct B register sets
#pragma unroll
            for (int t = 0; t < 4; ++t) {
                const __bf16* kpp = kcur + (t * 16 + l15) * 64 + ks * 32 + hi * 8;
                bm[t].u[0] = *(const u32x4*)(kpp);
                bm[t].u[1] = *(const u32x4*)(kpp + 16);
            }
#pragma unroll
            for (int t = 0; t < 4; ++t)        // independent, back-to-back
                c[t] = wmma_bf16(qa[ks].v, bm[t].v, c[t]);
        }

        // ---- scale + causal mask (diagonal tile only) ----------------------
        const bool diag = (kt == qt);
        const int  kb   = kt * 64;
#pragma unroll
        for (int t = 0; t < 4; ++t) {
            const int key = kb + t * 16 + l15;
#pragma unroll
            for (int j = 0; j < 8; ++j) {
                float sc = c[t][j] * 0.125f;              // 1/sqrt(64)
                if (diag) {
                    const int qr = qrowbase + j + 8 * hi;
                    sc = (key <= qr) ? sc : -1e30f;
                }
                c[t][j] = sc;
            }
        }

        // ---- online softmax (row reductions inside each 16-lane half) ------
#pragma unroll
        for (int j = 0; j < 8; ++j) {
            float rm = fmaxf(fmaxf(c[0][j], c[1][j]), fmaxf(c[2][j], c[3][j]));
#pragma unroll
            for (int o = 8; o >= 1; o >>= 1) rm = fmaxf(rm, __shfl_xor(rm, o, 32));
            const float mn    = fmaxf(m[j], rm);
            const float alpha = __expf(m[j] - mn);
            m[j] = mn;
            float rs = 0.0f;
#pragma unroll
            for (int t = 0; t < 4; ++t) {
                const float p = __expf(c[t][j] - mn);
                c[t][j] = p;
                rs += p;
            }
#pragma unroll
            for (int o = 8; o >= 1; o >>= 1) rs += __shfl_xor(rs, o, 32);
            lsum[j] = lsum[j] * alpha + rs;
#pragma unroll
            for (int t = 0; t < 4; ++t) acc[t][j] *= alpha;
        }

        // ---- relayout P: C layout -> A layout via per-wave LDS tile --------
#pragma unroll
        for (int t = 0; t < 4; ++t)
#pragma unroll
            for (int j = 0; j < 8; ++j)
                plds[wv][j + 8 * hi][t * 16 + l15] = (__bf16)c[t][j];

        // ---- out += P * V from LDS -----------------------------------------
#pragma unroll
        for (int ks = 0; ks < 2; ++ks) {
            ABf pa;
            const __bf16* pp = &plds[wv][l15][ks * 32 + off];
            pa.u[0] = *(const u32x4*)(pp);
            pa.u[1] = *(const u32x4*)(pp + 16);
            ABf bm[4];
#pragma unroll
            for (int t = 0; t < 4; ++t) {
                const __bf16* vp = vcur + (t * 16 + l15) * 64 + ks * 32 + hi * 8;
                bm[t].u[0] = *(const u32x4*)(vp);
                bm[t].u[1] = *(const u32x4*)(vp + 16);
            }
#pragma unroll
            for (int t = 0; t < 4; ++t)
                acc[t] = wmma_bf16(pa.v, bm[t].v, acc[t]);
        }

        wait_async0();     // next tile's async copies complete
        __syncthreads();   // all waves done reading cur + copies visible
    }

    // ---- normalize + nontemporal fp32 store (result never re-read) ---------
#pragma unroll
    for (int j = 0; j < 8; ++j) {
        const float inv = 1.0f / lsum[j];
        const int   qr  = qrowbase + j + 8 * hi;
        float* op = out + ((size_t)b * SS + qr) * HH;
#pragma unroll
        for (int t = 0; t < 4; ++t)
            __builtin_nontemporal_store(acc[t][j] * inv, op + t * 16 + l15);
    }
}

// ---------------------------------------------------------------------------
extern "C" void kernel_launch(void* const* d_in, const int* in_sizes, int n_in,
                              void* d_out, int out_size, void* d_ws, size_t ws_size,
                              hipStream_t stream) {
    const float* x  = (const float*)d_in[0];
    const float* Wq = (const float*)d_in[1];
    const float* bq = (const float*)d_in[2];
    const float* Wk = (const float*)d_in[3];
    const float* bk = (const float*)d_in[4];
    const float* Wv = (const float*)d_in[5];
    const float* bv = (const float*)d_in[6];
    float* out = (float*)d_out;

    const size_t qkv_bytes = (size_t)BB * SS * HH * 2;   // 2 MB each, bf16
    const size_t w_bytes   = (size_t)EE * HH * 2;        // 128 KB each, bf16
    char* wsb = (char*)d_ws;
    __bf16* qws  = (__bf16*)(wsb);
    __bf16* kws  = (__bf16*)(wsb + qkv_bytes);
    __bf16* vtws = (__bf16*)(wsb + 2 * qkv_bytes);
    __bf16* wtq  = (__bf16*)(wsb + 3 * qkv_bytes);
    __bf16* wtk  = (__bf16*)(wsb + 3 * qkv_bytes + w_bytes);
    __bf16* wtv  = (__bf16*)(wsb + 3 * qkv_bytes + 2 * w_bytes);

    dim3 g0(3 * EE * HH / 256), b0(256);
    wconv_kernel<<<g0, b0, 0, stream>>>(Wq, Wk, Wv, wtq, wtk, wtv);

    dim3 g1(BB * SS / 64, 3), b1(128);
    qkv_proj_kernel<<<g1, b1, 0, stream>>>(x, wtq, wtk, wtv, bq, bk, bv,
                                           qws, kws, vtws);

    dim3 g2(SS / 64, BB), b2(128);
    flash_attn_kernel<<<g2, b2, 0, stream>>>(qws, kws, vtws, out);
}
